// SchNet_18528488915074
// MI455X (gfx1250) — matrix operand order repaired and checked by reading, not compiled, not run on hardware
//
#include <hip/hip_runtime.h>
#include <hip/hip_bf16.h>

typedef __attribute__((ext_vector_type(16))) _Float16 v16h;
typedef __attribute__((ext_vector_type(4)))  _Float16 v4h;
typedef __attribute__((ext_vector_type(8)))  float    v8f;
typedef __attribute__((ext_vector_type(4)))  unsigned int u32x4;
typedef __attribute__((ext_vector_type(8)))  int          i32x8;
typedef __attribute__((ext_vector_type(4)))  int          i32x4;

#define SSP_SHIFT 0.6931471805599453f
#define CUTOFF_R  5.0f
#define PI_F      3.14159265358979323846f

// GEMM epilogue flags
#define GF_BIAS 1   // add bias[n]
#define GF_SSP  2   // shifted-softplus activation
#define GF_MULD 4   // multiply by ssp'(aux[m][n])  (aux holds activated values)
#define GF_ACC  8   // C += result  (else C = result)

__device__ __forceinline__ float sspf(float x) {
  float t = (x > 15.0f) ? x : log1pf(__expf(x));
  return t - SSP_SHIFT;
}
// ssp'(x) = sigmoid(x) = 1 - exp(-(y + shift)) where y = ssp(x)
__device__ __forceinline__ float ssp_deriv_from_y(float y) {
  return 1.0f - __expf(-(y + SSP_SHIFT));
}

// --------------------------------------------------------------------------
// Shared WMMA core helpers (fragment index maps; A and B share contiguous k = 16*half + s)
// A frag (16-bit 16x32): ka = half*16 + 2*v
// B frag:                kb = half*16 + 2*v
// C/D:                   m = 16*wave + r + 8*half, n = lane&15
// --------------------------------------------------------------------------

// ---------------------------------------------------------------------------
// f32-A WMMA GEMM (node-sized ops): C[M,16*NT] = epi( A[M,K](f32) @ Wt^T(f16) )
// Manual LDS staging with f32->f16 conversion. K%32==0, lda%4==0.
// ---------------------------------------------------------------------------
template <int NT>
__global__ __launch_bounds__(256) void gemm_wmma_f(
    const float* __restrict__ A, int lda,
    const _Float16* __restrict__ Wt, int ldw,
    const float* __restrict__ bias,
    const float* __restrict__ aux,
    float* __restrict__ C, int ldc,
    int M, int K, int flags)
{
  __shared__ _Float16 As[128][40];        // 32 cols + pad; 80B row = 16B aligned
  const int tid  = threadIdx.x;
  const int wave = tid >> 5;
  const int lane = tid & 31;
  const int half = lane >> 4;
  const int mn   = lane & 15;
  const int m0   = blockIdx.x * 128;

  v8f acc[NT];
#pragma unroll
  for (int t = 0; t < NT; ++t) acc[t] = (v8f){0.f,0.f,0.f,0.f,0.f,0.f,0.f,0.f};

  for (int k0 = 0; k0 < K; k0 += 32) {
    __syncthreads();
#pragma unroll
    for (int it = 0; it < 4; ++it) {
      int g  = it * 256 + tid;
      int r  = g >> 3;
      int cg = (g & 7) * 4;
      int gm = m0 + r; gm = (gm < M) ? gm : (M - 1);
      const float4 v = *(const float4*)(A + (size_t)gm * lda + (k0 + cg));
      v4h hv = { (_Float16)v.x, (_Float16)v.y, (_Float16)v.z, (_Float16)v.w };
      *(v4h*)&As[r][cg] = hv;
    }
    __syncthreads();

    v16h af;
    const _Float16* arow = &As[wave * 16 + mn][0];
#pragma unroll
    for (int v = 0; v < 8; ++v) {
      int ka = half * 16 + 2 * v;
      af[2 * v]     = arow[ka];
      af[2 * v + 1] = arow[ka + 1];
    }
#pragma unroll
    for (int t = 0; t < NT; ++t) {
      v16h bf;
      const _Float16* brow = Wt + (size_t)(t * 16 + mn) * ldw + k0;
#pragma unroll
      for (int v = 0; v < 8; ++v) {
        int kb = half * 16 + 2 * v;
        bf[2 * v]     = brow[kb];
        bf[2 * v + 1] = brow[kb + 1];
      }
      acc[t] = __builtin_amdgcn_wmma_f32_16x16x32_f16(
          false, af, false, bf, (short)0, acc[t], false, false);
    }
  }

#pragma unroll
  for (int t = 0; t < NT; ++t) {
    const int n = t * 16 + mn;
    const float b = (flags & GF_BIAS) ? bias[n] : 0.0f;
#pragma unroll
    for (int r = 0; r < 8; ++r) {
      int m = m0 + wave * 16 + r + 8 * half;
      if (m < M) {
        size_t o = (size_t)m * ldc + n;
        float x = acc[t][r] + b;
        if (flags & GF_SSP)  x = sspf(x);
        if (flags & GF_MULD) x *= ssp_deriv_from_y(aux[o]);
        if (flags & GF_ACC)  C[o] += x; else C[o] = x;
      }
    }
  }
}

// ---------------------------------------------------------------------------
// f16-A WMMA GEMM (edge-sized ops): A[M,K] f16 panel loaded once per block by
// the Tensor Data Mover (tensor_load_to_lds, TENSORcnt), with LDS row padding
// (K+8 halves, 272B/144B stride -> bank-conflict free, 16B aligned).
// C16: store f16 output (next consumer is a WMMA A-operand anyway).
// ---------------------------------------------------------------------------
template <int NT, bool C16>
__global__ __launch_bounds__(256) void gemm_wmma_h(
    const _Float16* __restrict__ A, int lda,       // in halves
    const _Float16* __restrict__ Wt, int ldw,
    const float* __restrict__ bias,
    const _Float16* __restrict__ aux,
    void* __restrict__ Cv, int ldc,
    int M, int K, int flags)
{
  __shared__ _Float16 As[128][136];                // up to K=128 (+8 pad)
  const int tid  = threadIdx.x;
  const int wave = tid >> 5;
  const int lane = tid & 31;
  const int half = lane >> 4;
  const int mn   = lane & 15;
  const int m0   = blockIdx.x * 128;
  const int ldsK = K + 8;                          // used row stride in halves

  // --- TDM: one strided 2D tile (128 x K halves) -> LDS, issued by wave 0 ---
  if (tid < 32) {
    unsigned long long ga =
        (unsigned long long)(const void*)(A + (size_t)m0 * lda);
    unsigned ldsaddr = (unsigned)(unsigned long long)(const void*)&As[0][0];
    int rows = M - m0; if (rows > 128) rows = 128;
    unsigned pad_int = (K == 128) ? 5u : 4u;       // 64 / 32 DWORD interval
    unsigned pad_dw  = (unsigned)((136 - K) >> 1); // pad in DWORDs
    u32x4 g0;
    g0[0] = 1u;                                    // count=1, user descriptor
    g0[1] = ldsaddr;                               // lds_addr
    g0[2] = (unsigned)(ga & 0xffffffffu);          // global_addr[31:0]
    g0[3] = (unsigned)((ga >> 32) & 0x01ffffffu) | (2u << 30);  // type=2
    i32x8 g1;
    g1[0] = (int)((1u << 16) | (1u << 20) | (pad_int << 22) | ((pad_dw - 1u) << 25));
    g1[1] = (int)((unsigned)K << 16);              // tensor_dim0[15:0] = K
    g1[2] = (int)((unsigned)rows << 16);           // tensor_dim1[15:0] = rows
    g1[3] = (int)((unsigned)K << 16);              // tile_dim0 = K
    g1[4] = 128;                                   // tile_dim1 = 128 rows
    g1[5] = lda;                                   // tensor_dim0_stride
    g1[6] = 0;
    g1[7] = 0;
    i32x4 gz4 = {0, 0, 0, 0};
    i32x8 gz8 = {0, 0, 0, 0, 0, 0, 0, 0};
    __builtin_amdgcn_tensor_load_to_lds(g0, g1, gz4, gz4, gz8, 0);
    __builtin_amdgcn_s_wait_tensorcnt(0);
  }
  __syncthreads();

  v8f acc[NT];
#pragma unroll
  for (int t = 0; t < NT; ++t) acc[t] = (v8f){0.f,0.f,0.f,0.f,0.f,0.f,0.f,0.f};

  const _Float16* arow = &As[0][0] + (size_t)(wave * 16 + mn) * ldsK;
  for (int k0 = 0; k0 < K; k0 += 32) {
    v16h af;
#pragma unroll
    for (int v = 0; v < 8; ++v) {
      int ka = k0 + half * 16 + 2 * v;
      af[2 * v]     = arow[ka];
      af[2 * v + 1] = arow[ka + 1];
    }
#pragma unroll
    for (int t = 0; t < NT; ++t) {
      v16h bf;
      const _Float16* brow = Wt + (size_t)(t * 16 + mn) * ldw + k0;
#pragma unroll
      for (int v = 0; v < 8; ++v) {
        int kb = half * 16 + 2 * v;
        bf[2 * v]     = brow[kb];
        bf[2 * v + 1] = brow[kb + 1];
      }
      acc[t] = __builtin_amdgcn_wmma_f32_16x16x32_f16(
          false, af, false, bf, (short)0, acc[t], false, false);
    }
  }

#pragma unroll
  for (int t = 0; t < NT; ++t) {
    const int n = t * 16 + mn;
    const float b = (flags & GF_BIAS) ? bias[n] : 0.0f;
#pragma unroll
    for (int r = 0; r < 8; ++r) {
      int m = m0 + wave * 16 + r + 8 * half;
      if (m < M) {
        size_t o = (size_t)m * ldc + n;
        float x = acc[t][r] + b;
        if (flags & GF_SSP)  x = sspf(x);
        if (flags & GF_MULD) x *= ssp_deriv_from_y((float)aux[o]);
        if (C16) {
          ((_Float16*)Cv)[o] = (_Float16)x;
        } else {
          float* C = (float*)Cv;
          if (flags & GF_ACC) C[o] += x; else C[o] = x;
        }
      }
    }
  }
}

// ---------------------------------------------------------------------------
// Weight conversion: f32 W[rows][cols] -> f16 D[drows][dK]
// transpose=1: D[n][k] = W[k][n] (zero pad); transpose=0: D[n][k] = W[n][k]
// ---------------------------------------------------------------------------
__global__ void convw_k(const float* __restrict__ W, int rows, int cols,
                        _Float16* __restrict__ D, int drows, int dK, int transpose)
{
  int i = blockIdx.x * blockDim.x + threadIdx.x;
  if (i >= drows * dK) return;
  int n = i / dK, k = i % dK;
  float v = 0.0f;
  if (transpose) { if (k < rows && n < cols) v = W[(size_t)k * cols + n]; }
  else           { if (n < rows && k < cols) v = W[(size_t)n * cols + k]; }
  D[i] = (_Float16)v;
}

__global__ void zero_k(float* __restrict__ p, size_t n) {
  for (size_t i = blockIdx.x * (size_t)blockDim.x + threadIdx.x; i < n;
       i += (size_t)gridDim.x * blockDim.x) p[i] = 0.0f;
}

__global__ void embed_k(const int* __restrict__ z, const float* __restrict__ emb,
                        float* __restrict__ h0, int N) {
  int i = blockIdx.x * blockDim.x + threadIdx.x;
  if (i >= N * 128) return;
  int n = i >> 7, j = i & 127;
  h0[i] = emb[(size_t)z[n] * 128 + j];
}

// distances, Gaussian smearing (f16, padded to 64 cols), cosine cutoff
__global__ void edge_geom_k(const float* __restrict__ pos, const int* __restrict__ ei,
                            int E, float* __restrict__ diff, float* __restrict__ d,
                            float* __restrict__ c, _Float16* __restrict__ attr)
{
  int e = blockIdx.x * blockDim.x + threadIdx.x;
  if (e >= E) return;
  int r = ei[e], cl = ei[E + e];
  float dx = pos[r * 3 + 0] - pos[cl * 3 + 0];
  float dy = pos[r * 3 + 1] - pos[cl * 3 + 1];
  float dz = pos[r * 3 + 2] - pos[cl * 3 + 2];
  float dist = sqrtf(dx * dx + dy * dy + dz * dz);
  diff[e * 3 + 0] = dx; diff[e * 3 + 1] = dy; diff[e * 3 + 2] = dz;
  d[e] = dist;
  c[e] = 0.5f * (__cosf(dist * (PI_F / CUTOFF_R)) + 1.0f);
  const float delta = CUTOFF_R / 49.0f;
  const float gamma = 0.5f / (delta * delta);
#pragma unroll 2
  for (int g = 0; g < 64; ++g) {
    float t = dist - (float)g * delta;
    attr[(size_t)e * 64 + g] = (_Float16)((g < 50) ? __expf(-gamma * t * t) : 0.0f);
  }
}

// msg = hh[col] * (c*filt); agg[row] += msg.  One wave32 per edge.
__global__ __launch_bounds__(256) void msg_scatter_k(
    const int* __restrict__ ei, const float* __restrict__ c,
    const _Float16* __restrict__ filt, const float* __restrict__ hh,
    float* __restrict__ agg, int E)
{
  int w = (blockIdx.x * blockDim.x + threadIdx.x) >> 5;
  int lane = threadIdx.x & 31;
  if (w >= E) return;
  int r = ei[w], cl = ei[E + w];
  float ce = c[w];
  v4h f4 = *(const v4h*)(filt + (size_t)w * 128 + lane * 4);
  float4 h4 = *(const float4*)(hh + (size_t)cl * 128 + lane * 4);
  float* ap = agg + (size_t)r * 128 + lane * 4;
  unsafeAtomicAdd(ap + 0, h4.x * ce * (float)f4[0]);
  unsafeAtomicAdd(ap + 1, h4.y * ce * (float)f4[1]);
  unsafeAtomicAdd(ap + 2, h4.z * ce * (float)f4[2]);
  unsafeAtomicAdd(ap + 3, h4.w * ce * (float)f4[3]);
}

// backward of message/scatter: g_filt (f16), g_hh (scatter f32), g_c (+=)
__global__ __launch_bounds__(256) void edge_bwd_k(
    const int* __restrict__ ei, const float* __restrict__ c,
    const float* __restrict__ g_agg, const float* __restrict__ hh,
    const _Float16* __restrict__ filt, _Float16* __restrict__ g_filt,
    float* __restrict__ g_hh, float* __restrict__ g_c, int E)
{
  int w = (blockIdx.x * blockDim.x + threadIdx.x) >> 5;
  int lane = threadIdx.x & 31;
  if (w >= E) return;
  int r = ei[w], cl = ei[E + w];
  float ce = c[w];
  float4 ga = *(const float4*)(g_agg + (size_t)r * 128 + lane * 4);
  float4 hc = *(const float4*)(hh + (size_t)cl * 128 + lane * 4);
  v4h ft4 = *(const v4h*)(filt + (size_t)w * 128 + lane * 4);
  float fx = (float)ft4[0], fy = (float)ft4[1], fz = (float)ft4[2], fw = (float)ft4[3];
  v4h gf = { (_Float16)(ga.x * hc.x * ce), (_Float16)(ga.y * hc.y * ce),
             (_Float16)(ga.z * hc.z * ce), (_Float16)(ga.w * hc.w * ce) };
  *(v4h*)(g_filt + (size_t)w * 128 + lane * 4) = gf;
  float* gp = g_hh + (size_t)cl * 128 + lane * 4;
  unsafeAtomicAdd(gp + 0, ga.x * ce * fx);
  unsafeAtomicAdd(gp + 1, ga.y * ce * fy);
  unsafeAtomicAdd(gp + 2, ga.z * ce * fz);
  unsafeAtomicAdd(gp + 3, ga.w * ce * fw);
  float p = ga.x * hc.x * fx + ga.y * hc.y * fy + ga.z * hc.z * fz + ga.w * hc.w * fw;
#pragma unroll
  for (int o = 16; o > 0; o >>= 1) p += __shfl_xor(p, o, 32);
  if (lane == 0) g_c[w] += p;
}

// readout: y_n = dot(s1_n, out_w2) + b2; energy += y_n
__global__ void energy_k(const float* __restrict__ o1s, const float* __restrict__ w2,
                         const float* __restrict__ b2, float* __restrict__ energy, int N)
{
  int n = blockIdx.x * blockDim.x + threadIdx.x;
  if (n >= N) return;
  float y = b2[0];
#pragma unroll 8
  for (int j = 0; j < 64; ++j) y += o1s[(size_t)n * 64 + j] * w2[j];
  unsafeAtomicAdd(energy, y);
}

__global__ void grad_o1_k(const float* __restrict__ o1s, const float* __restrict__ w2,
                          float* __restrict__ g_o1, int N)
{
  int i = blockIdx.x * blockDim.x + threadIdx.x;
  if (i >= N * 64) return;
  g_o1[i] = w2[i & 63] * ssp_deriv_from_y(o1s[i]);
}

// edge force accumulation: g_d -> atomic forces (forces = -grad_pos)
__global__ void force_k(const int* __restrict__ ei, const float* __restrict__ diff,
                        const float* __restrict__ d, const _Float16* __restrict__ attr,
                        const float* __restrict__ g_attr, const float* __restrict__ g_c,
                        float* __restrict__ forces, int E)
{
  int e = blockIdx.x * blockDim.x + threadIdx.x;
  if (e >= E) return;
  float dist = d[e];
  const float delta = CUTOFF_R / 49.0f;
  const float gamma = 0.5f / (delta * delta);
  float gd = 0.0f;
#pragma unroll 5
  for (int g = 0; g < 50; ++g) {
    float t = dist - (float)g * delta;
    gd += g_attr[(size_t)e * 64 + g] * (-2.0f * gamma * t) * (float)attr[(size_t)e * 64 + g];
  }
  gd += g_c[e] * (-0.5f * (PI_F / CUTOFF_R)) * __sinf(dist * (PI_F / CUTOFF_R));
  float s = gd / dist;
  int r = ei[e], cl = ei[E + e];
#pragma unroll
  for (int j = 0; j < 3; ++j) {
    float v = s * diff[e * 3 + j];
    unsafeAtomicAdd(&forces[(size_t)r * 3 + j], -v);
    unsafeAtomicAdd(&forces[(size_t)cl * 3 + j], v);
  }
}

// ---------------------------------------------------------------------------
extern "C" void kernel_launch(void* const* d_in, const int* in_sizes, int n_in,
                              void* d_out, int out_size, void* d_ws, size_t ws_size,
                              hipStream_t stream)
{
  const int H = 128, F = 128, L = 6;
  const int N = in_sizes[0];
  const int E = in_sizes[2] / 2;

  const int*   z      = (const int*)d_in[0];
  const float* pos    = (const float*)d_in[1];
  const int*   ei     = (const int*)d_in[2];
  const float* emb    = (const float*)d_in[3];
  const float* mlp_w1 = (const float*)d_in[4];
  const float* mlp_b1 = (const float*)d_in[5];
  const float* mlp_w2 = (const float*)d_in[6];
  const float* mlp_b2 = (const float*)d_in[7];
  const float* cf_w1  = (const float*)d_in[8];
  const float* cf_w2  = (const float*)d_in[9];
  const float* cf_b2  = (const float*)d_in[10];
  const float* int_w  = (const float*)d_in[11];
  const float* int_b  = (const float*)d_in[12];
  const float* out_w1 = (const float*)d_in[13];
  const float* out_b1 = (const float*)d_in[14];
  const float* out_w2 = (const float*)d_in[15];
  const float* out_b2 = (const float*)d_in[16];
  float* out = (float*)d_out;

  size_t off = 0;
  auto allocB = [&](size_t bytes) -> char* {
    off = (off + 255) & ~(size_t)255;
    char* p = (char*)d_ws + off;
    off += bytes;
    return p;
  };
  auto allocF = [&](size_t n) { return (float*)allocB(n * 4); };
  auto allocH = [&](size_t n) { return (_Float16*)allocB(n * 2); };

  _Float16 *w1T[6], *w1C[6], *w2T[6], *w2C[6], *c1T[6], *c1C[6],
           *c2T[6], *c2C[6], *iwT[6], *iwC[6];
  for (int i = 0; i < L; ++i) {
    w1T[i] = allocH(128 * 64);  w1C[i] = allocH(64 * 128);
    w2T[i] = allocH(128 * 128); w2C[i] = allocH(128 * 128);
    c1T[i] = allocH(128 * 128); c1C[i] = allocH(128 * 128);
    c2T[i] = allocH(128 * 128); c2C[i] = allocH(128 * 128);
    iwT[i] = allocH(128 * 128); iwC[i] = allocH(128 * 128);
  }
  _Float16* ow1T = allocH(64 * 128);
  _Float16* ow1C = allocH(128 * 64);

  _Float16* attr  = allocH((size_t)E * 64);   // f16, cols 50..63 zero
  float* dist     = allocF(E);
  float* cutf     = allocF(E);
  float* diff     = allocF((size_t)E * 3);
  float* hhist    = allocF((size_t)(L + 1) * N * H);
  float* shist    = allocF((size_t)L * N * H);
  float* hhhist   = allocF((size_t)L * N * F);
  float* agg      = allocF((size_t)N * F);
  float* o1s      = allocF((size_t)N * 64);
  float* go1      = allocF((size_t)N * 64);
  float* gh       = allocF((size_t)N * H);
  float* gu       = allocF((size_t)N * H);
  float* gagg     = allocF((size_t)N * F);
  float* ghh      = allocF((size_t)N * F);
  _Float16* Bt1   = allocH((size_t)E * F);    // f16 edge intermediates
  _Float16* Bfilt = allocH((size_t)E * F);
  _Float16* gfilt = allocH((size_t)E * F);
  float* gattr    = allocF((size_t)E * 64);
  float* gc       = allocF(E);
  (void)ws_size; (void)n_in;

  auto zero = [&](float* p, size_t n) {
    int blocks = (int)(((n + 255) / 256 < 8192) ? (n + 255) / 256 : 8192);
    zero_k<<<blocks, 256, 0, stream>>>(p, n);
  };
  auto conv = [&](const float* W, int r, int c, _Float16* D, int dr, int dk, int tr) {
    int tot = dr * dk;
    convw_k<<<(tot + 255) / 256, 256, 0, stream>>>(W, r, c, D, dr, dk, tr);
  };
  auto gemmF = [&](const float* A, int lda, const _Float16* Wt, int ldw,
                   const float* bias, const float* aux, float* C, int ldc,
                   int M, int K, int Ncols, int flags) {
    dim3 grid((M + 127) / 128, 1);
    if (Ncols == 128)
      gemm_wmma_f<8><<<grid, 256, 0, stream>>>(A, lda, Wt, ldw, bias, aux, C, ldc, M, K, flags);
    else
      gemm_wmma_f<4><<<grid, 256, 0, stream>>>(A, lda, Wt, ldw, bias, aux, C, ldc, M, K, flags);
  };
  auto gemmH16 = [&](const _Float16* A, int lda, const _Float16* Wt, int ldw,
                     const float* bias, const _Float16* aux, _Float16* C, int ldc,
                     int M, int K, int flags) {     // Ncols = 128, f16 out
    dim3 grid((M + 127) / 128, 1);
    gemm_wmma_h<8, true><<<grid, 256, 0, stream>>>(A, lda, Wt, ldw, bias, aux, C, ldc, M, K, flags);
  };
  auto gemmH32 = [&](const _Float16* A, int lda, const _Float16* Wt, int ldw,
                     float* C, int ldc, int M, int K, int flags) { // Ncols = 64, f32 out
    dim3 grid((M + 127) / 128, 1);
    gemm_wmma_h<4, false><<<grid, 256, 0, stream>>>(A, lda, Wt, ldw, nullptr, nullptr,
                                                    C, ldc, M, K, flags);
  };

  // ---- weight prep -------------------------------------------------------
  for (int i = 0; i < L; ++i) {
    conv(mlp_w1 + (size_t)i * 50 * F, 50, F, w1T[i], 128, 64, 1);
    conv(mlp_w1 + (size_t)i * 50 * F, 50, F, w1C[i], 64, 128, 0);
    conv(mlp_w2 + (size_t)i * F * F, F, F, w2T[i], 128, 128, 1);
    conv(mlp_w2 + (size_t)i * F * F, F, F, w2C[i], 128, 128, 0);
    conv(cf_w1 + (size_t)i * H * F, H, F, c1T[i], 128, 128, 1);
    conv(cf_w1 + (size_t)i * H * F, H, F, c1C[i], 128, 128, 0);
    conv(cf_w2 + (size_t)i * F * H, F, H, c2T[i], 128, 128, 1);
    conv(cf_w2 + (size_t)i * F * H, F, H, c2C[i], 128, 128, 0);
    conv(int_w + (size_t)i * H * H, H, H, iwT[i], 128, 128, 1);
    conv(int_w + (size_t)i * H * H, H, H, iwC[i], 128, 128, 0);
  }
  conv(out_w1, H, 64, ow1T, 64, 128, 1);
  conv(out_w1, H, 64, ow1C, 128, 64, 0);

  // ---- forward -----------------------------------------------------------
  zero(out, (size_t)out_size);
  edge_geom_k<<<(E + 255) / 256, 256, 0, stream>>>(pos, ei, E, diff, dist, cutf, attr);
  embed_k<<<(N * H + 255) / 256, 256, 0, stream>>>(z, emb, hhist, N);

  for (int i = 0; i < L; ++i) {
    float* h_i  = hhist + (size_t)i * N * H;
    float* h_o  = hhist + (size_t)(i + 1) * N * H;
    float* s_i  = shist + (size_t)i * N * H;
    float* hh_i = hhhist + (size_t)i * N * F;
    // t1 = ssp(attr @ W1 + b1)            [TDM-staged f16 A, f16 out]
    gemmH16(attr, 64, w1T[i], 64, mlp_b1 + (size_t)i * F, nullptr, Bt1, 128, E, 64,
            GF_BIAS | GF_SSP);
    // filt = t1 @ W2 + b2
    gemmH16(Bt1, 128, w2T[i], 128, mlp_b2 + (size_t)i * F, nullptr, Bfilt, 128, E, 128,
            GF_BIAS);
    // hh = h @ cf_w1
    gemmF(h_i, 128, c1T[i], 128, nullptr, nullptr, hh_i, 128, N, 128, 128, 0);
    // agg = scatter(row, hh[col] * c * filt)
    zero(agg, (size_t)N * F);
    msg_scatter_k<<<(E + 7) / 8, 256, 0, stream>>>(ei, cutf, Bfilt, hh_i, agg, E);
    // s = ssp(agg @ cf_w2 + cf_b2)
    gemmF(agg, 128, c2T[i], 128, cf_b2 + (size_t)i * H, nullptr, s_i, 128, N, 128, 128,
          GF_BIAS | GF_SSP);
    // h_out = s @ int_w + int_b
    gemmF(s_i, 128, iwT[i], 128, int_b + (size_t)i * H, nullptr, h_o, 128, N, 128, 128,
          GF_BIAS);
  }
  gemmF(hhist + (size_t)L * N * H, 128, ow1T, 128, out_b1, nullptr, o1s, 64, N, 128, 64,
        GF_BIAS | GF_SSP);
  energy_k<<<(N + 255) / 256, 256, 0, stream>>>(o1s, out_w2, out_b2, out + (size_t)N * 3, N);

  // ---- backward ----------------------------------------------------------
  grad_o1_k<<<(N * 64 + 255) / 256, 256, 0, stream>>>(o1s, out_w2, go1, N);
  gemmF(go1, 64, ow1C, 64, nullptr, nullptr, gh, 128, N, 64, 128, 0);
  zero(gattr, (size_t)E * 64);
  zero(gc, (size_t)E);

  for (int i = L - 1; i >= 0; --i) {
    float* s_i  = shist + (size_t)i * N * H;
    float* hh_i = hhhist + (size_t)i * N * F;
    // g_u = (g_h @ int_w^T) * ssp'(u)
    gemmF(gh, 128, iwC[i], 128, nullptr, s_i, gu, 128, N, 128, 128, GF_MULD);
    // g_agg = g_u @ cf_w2^T
    gemmF(gu, 128, c2C[i], 128, nullptr, nullptr, gagg, 128, N, 128, 128, 0);
    // recompute t1, filt (WMMA flops << keeping 6 layers of [E,F] resident)
    gemmH16(attr, 64, w1T[i], 64, mlp_b1 + (size_t)i * F, nullptr, Bt1, 128, E, 64,
            GF_BIAS | GF_SSP);
    gemmH16(Bt1, 128, w2T[i], 128, mlp_b2 + (size_t)i * F, nullptr, Bfilt, 128, E, 128,
            GF_BIAS);
    // edge backward
    zero(ghh, (size_t)N * F);
    edge_bwd_k<<<(E + 7) / 8, 256, 0, stream>>>(ei, cutf, gagg, hh_i, Bfilt, gfilt, ghh, gc, E);
    // g_h(prev) = g_hh @ cf_w1^T
    gemmF(ghh, 128, c1C[i], 128, nullptr, nullptr, gh, 128, N, 128, 128, 0);
    // g_pre1 = (g_filt @ W2^T) * ssp'(t1)  -> Bfilt (f16)
    gemmH16(gfilt, 128, w2C[i], 128, nullptr, Bt1, Bfilt, 128, E, 128, GF_MULD);
    // g_attr += g_pre1 @ W1^T  (f32 accumulate)
    gemmH32(Bfilt, 128, w1C[i], 128, gattr, 64, E, 128, GF_ACC);
  }

  force_k<<<(E + 255) / 256, 256, 0, stream>>>(ei, diff, dist, attr, gattr, gc, out, E);
}
